// MultiHeadAttention_28570122453895
// MI455X (gfx1250) — matrix operand order repaired and checked by reading, not compiled, not run on hardware
//
#include <hip/hip_runtime.h>

// ---------------- types ----------------
typedef __attribute__((ext_vector_type(16))) _Float16 v16h;
typedef __attribute__((ext_vector_type(8)))  _Float16 h8;
typedef __attribute__((ext_vector_type(2)))  _Float16 h2;
typedef __attribute__((ext_vector_type(8)))  float    v8f;

// ---------------- problem constants ----------------
constexpr int   BB   = 2;
constexpr int   SS   = 2048;
constexpr int   HH   = 16;
constexpr int   DD   = 64;
constexpr int   DM   = 1024;
constexpr float SCALE = 0.125f;     // 1/sqrt(64)

// LDS tile stride (in halves); multiple of 8 so fragment loads are 16B aligned
constexpr int TS = 72;

// ---------------- WMMA helpers ----------------
__device__ __forceinline__ v8f wmma_f16(v16h a, v16h b, v8f c) {
  return __builtin_amdgcn_wmma_f32_16x16x32_f16(false, a, false, b, (short)0, c, false, false);
}

// A-matrix 16x32 f16 fragment (ISA 7.12.2): lane<16: row=lane, K = {blo..blo+7, blo+16..blo+23}
// with blo = kbase (+8 for lanes 16-31).
__device__ __forceinline__ v16h load_a_frag(const _Float16* base, int stride, int lane, int kbase) {
  int m   = lane & 15;
  int blo = kbase + ((lane >> 4) << 3);
  const _Float16* p = base + m * stride + blo;
  h8 lo = *(const h8*)(p);
  h8 hi = *(const h8*)(p + 16);
  return __builtin_shufflevector(lo, hi, 0,1,2,3,4,5,6,7,8,9,10,11,12,13,14,15);
}

// B-matrix 32x16 f16 fragment: lane holds column n=lane&15, K = {blo..blo+7, blo+16..blo+23}, blo = kbase + 8*(lane>>4).
// `base` points at an LDS tile laid out [n][k] (i.e. B transposed, so reads are contiguous).
__device__ __forceinline__ v16h load_b_frag(const _Float16* base, int stride, int lane, int kbase) {
  int n  = lane & 15;
  int k0 = kbase + ((lane >> 4) << 3);
  const _Float16* p = base + n * stride + k0;
  h8 lo = *(const h8*)(p);
  h8 hi = *(const h8*)(p + 16);
  return __builtin_shufflevector(lo, hi, 0,1,2,3,4,5,6,7,8,9,10,11,12,13,14,15);
}

__device__ __forceinline__ float redmax16(float v) {
  #pragma unroll
  for (int m = 8; m >= 1; m >>= 1) v = fmaxf(v, __shfl_xor(v, m, 32));
  return v;
}
__device__ __forceinline__ float redsum16(float v) {
  #pragma unroll
  for (int m = 8; m >= 1; m >>= 1) v += __shfl_xor(v, m, 32);
  return v;
}

// ============================================================================
// Kernel 1: C[M,N] = A[M,K] @ W[K,N] + bias   (fp32 in/out, f16 WMMA compute)
// block = 128 threads (4 waves); block tile 64x64; wave tile 16x64; K-chunk 64.
// ============================================================================
__global__ __launch_bounds__(128)
void gemm_bias(const float* __restrict__ A, const float* __restrict__ W,
               const float* __restrict__ bias, float* __restrict__ C,
               int M, int N, int K) {
  __shared__ __align__(32) _Float16 As[64 * TS];
  __shared__ __align__(32) _Float16 Wt[64 * TS];

  int t = threadIdx.x, lane = t & 31, w = t >> 5, grp = lane >> 4, ln = lane & 15;
  int m0 = blockIdx.y * 64, n0 = blockIdx.x * 64;

  v8f acc[4];
  #pragma unroll
  for (int j = 0; j < 4; ++j) acc[j] = {0.f,0.f,0.f,0.f,0.f,0.f,0.f,0.f};

  for (int kk = 0; kk < K; kk += 64) {
    __syncthreads();
    { // stage A tile 64x64 (f32 -> f16), row-major; float4 (b128) global loads
      int row = t >> 1, c0 = (t & 1) * 32;
      const float4* src = (const float4*)(A + (size_t)(m0 + row) * K + kk + c0);
      _Float16* dst = As + row * TS + c0;
      #pragma unroll
      for (int i = 0; i < 8; ++i) {
        float4 f = src[i];
        dst[4*i+0] = (_Float16)f.x; dst[4*i+1] = (_Float16)f.y;
        dst[4*i+2] = (_Float16)f.z; dst[4*i+3] = (_Float16)f.w;
      }
    }
    { // stage W tile 64x64 transposed -> Wt[n][k]; pack (k,k+1) pairs as b32 stores
      int kp = (t >> 2) * 2, nseg = (t & 3) * 16;
      const float4* s0 = (const float4*)(W + (size_t)(kk + kp)     * N + n0 + nseg);
      const float4* s1 = (const float4*)(W + (size_t)(kk + kp + 1) * N + n0 + nseg);
      #pragma unroll
      for (int i = 0; i < 4; ++i) {
        float4 fa = s0[i], fb = s1[i];
        *(h2*)(Wt + (nseg + 4*i + 0) * TS + kp) = h2{(_Float16)fa.x, (_Float16)fb.x};
        *(h2*)(Wt + (nseg + 4*i + 1) * TS + kp) = h2{(_Float16)fa.y, (_Float16)fb.y};
        *(h2*)(Wt + (nseg + 4*i + 2) * TS + kp) = h2{(_Float16)fa.z, (_Float16)fb.z};
        *(h2*)(Wt + (nseg + 4*i + 3) * TS + kp) = h2{(_Float16)fa.w, (_Float16)fb.w};
      }
    }
    __syncthreads();
    v16h a0 = load_a_frag(As + (w * 16) * TS, TS, lane, 0);
    v16h a1 = load_a_frag(As + (w * 16) * TS, TS, lane, 32);
    #pragma unroll
    for (int j = 0; j < 4; ++j) {
      v16h b0 = load_b_frag(Wt + (j * 16) * TS, TS, lane, 0);
      v16h b1 = load_b_frag(Wt + (j * 16) * TS, TS, lane, 32);
      acc[j] = wmma_f16(a0, b0, acc[j]);
      acc[j] = wmma_f16(a1, b1, acc[j]);
    }
  }

  #pragma unroll
  for (int j = 0; j < 4; ++j) {
    int col = n0 + j * 16 + ln;
    float bv = bias[col];
    #pragma unroll
    for (int r = 0; r < 8; ++r) {
      int row = m0 + w * 16 + grp * 8 + r;
      C[(size_t)row * N + col] = acc[j][r] + bv;
    }
  }
}

// ============================================================================
// Kernel 2a: collapse the 512-dim gate MLP into bilinear forms.
// A_c[d][e] = sum_g Wgk[d][g]*Wm[g][c]*Wgq[e][g];  u_c = Wgk@(bgq*Wm_c);
// v_c = Wgq@(bgk*Wm_c);  w_c = sum_g bgk*bgq*Wm_c + bm[c].
// ============================================================================
__global__ __launch_bounds__(64)
void gate_precompute(const float* __restrict__ Wgk, const float* __restrict__ bgk,
                     const float* __restrict__ Wgq, const float* __restrict__ bgq,
                     const float* __restrict__ Wm,  const float* __restrict__ bm,
                     float* __restrict__ PA, float* __restrict__ U,
                     float* __restrict__ V,  float* __restrict__ Wc) {
  int d = blockIdx.x, c = blockIdx.y, e = threadIdx.x;
  float acc = 0.f;
  for (int g = 0; g < 512; ++g)
    acc += Wgk[d * 512 + g] * Wm[g * 2 + c] * Wgq[e * 512 + g];
  PA[((size_t)c * 64 + d) * 64 + e] = acc;
  if (d == 0) {
    float u = 0.f, vv = 0.f;
    for (int g = 0; g < 512; ++g) {
      float wm = Wm[g * 2 + c];
      u  += Wgk[e * 512 + g] * bgq[g] * wm;
      vv += Wgq[e * 512 + g] * bgk[g] * wm;
    }
    U[c * 64 + e] = u;
    V[c * 64 + e] = vv;
    if (e == 0) {
      float ww = bm[c];
      for (int g = 0; g < 512; ++g) ww += bgk[g] * bgq[g] * Wm[g * 2 + c];
      Wc[c] = ww;
    }
  }
}

// ============================================================================
// Kernel 2b: gate M[b,h,s,c] = sigmoid( kh^T A_c qh + u_c.kh + v_c.qh + w_c )
// one thread per (b,h,s); A_c in LDS (32KB), qh/kh kept in registers.
// ============================================================================
__global__ __launch_bounds__(256)
void gate_kernel(const float* __restrict__ Qp, const float* __restrict__ Kp,
                 const float* __restrict__ PA, const float* __restrict__ U,
                 const float* __restrict__ V,  const float* __restrict__ Wc,
                 float* __restrict__ Mg) {
  __shared__ float Asm[2 * 64 * 64];
  __shared__ float Us[128], Vs[128], Ws[2];
  for (int i = threadIdx.x; i < 8192; i += blockDim.x) Asm[i] = PA[i];
  for (int i = threadIdx.x; i < 128;  i += blockDim.x) { Us[i] = U[i]; Vs[i] = V[i]; }
  if (threadIdx.x < 2) Ws[threadIdx.x] = Wc[threadIdx.x];
  __syncthreads();

  int r = blockIdx.x * blockDim.x + threadIdx.x;   // 0..65535
  int s = r & (SS - 1);
  int h = (r >> 11) & (HH - 1);
  int b = r >> 15;
  const float4* qh4 = (const float4*)(Qp + ((size_t)(b * SS + s)) * DM + h * DD);
  const float4* kh4 = (const float4*)(Kp + ((size_t)(b * SS + s)) * DM + h * DD);

  float q[64], kr[64];
  #pragma unroll
  for (int e = 0; e < 16; ++e) {
    float4 fq = qh4[e], fk = kh4[e];
    q[4*e+0] = fq.x; q[4*e+1] = fq.y; q[4*e+2] = fq.z; q[4*e+3] = fq.w;
    kr[4*e+0] = fk.x; kr[4*e+1] = fk.y; kr[4*e+2] = fk.z; kr[4*e+3] = fk.w;
  }

  float out[2];
  #pragma unroll
  for (int c = 0; c < 2; ++c) {
    const float* A = Asm + c * 4096;
    float acc = Ws[c];
    for (int d = 0; d < 64; ++d) {
      float inner = Us[c * 64 + d];
      const float* Ar = A + d * 64;
      #pragma unroll
      for (int e = 0; e < 64; ++e) inner += Ar[e] * q[e];
      acc += kr[d] * inner;
    }
    #pragma unroll
    for (int e = 0; e < 64; ++e) acc += Vs[c * 64 + e] * q[e];
    out[c] = 1.f / (1.f + __expf(-acc));
  }
  size_t idx = ((size_t)(b * HH + h) * SS + s) * 2;
  Mg[idx] = out[0];
  Mg[idx + 1] = out[1];
}

// ----- shared staging helper: gated (B,S,DM)-slice tile -> f16 LDS [64][TS] -----
__device__ __forceinline__ void stage_gated_tile(const float* __restrict__ src_base,
                                                 const float* __restrict__ Mg,
                                                 int bh, int b, int h, int row0,
                                                 int t, int which, _Float16* dst_tile) {
  int row = t >> 1, d0 = (t & 1) * 32;
  int sr = row0 + row;
  float g = Mg[(((size_t)bh) * SS + sr) * 2 + which];
  const float4* src = (const float4*)(src_base + ((size_t)(b * SS + sr)) * DM + h * DD + d0);
  _Float16* dst = dst_tile + row * TS + d0;
  #pragma unroll
  for (int i = 0; i < 8; ++i) {
    float4 f = src[i];
    dst[4*i+0] = (_Float16)(f.x * g); dst[4*i+1] = (_Float16)(f.y * g);
    dst[4*i+2] = (_Float16)(f.z * g); dst[4*i+3] = (_Float16)(f.w * g);
  }
}

// ============================================================================
// Kernel 3a: per-row softmax stats (online max / sumexp), nothing stored but
// 2 floats per row. block=128 (4 waves) handles 64 q rows of one (b,h).
// ============================================================================
__global__ __launch_bounds__(128)
void attn_stats(const float* __restrict__ Qp, const float* __restrict__ Kp,
                const float* __restrict__ Mg, const float* __restrict__ mask,
                float* __restrict__ rowM, float* __restrict__ rowL) {
  int bh = blockIdx.x;
  int b = bh >> 4, h = bh & 15;
  int qbase = blockIdx.y * 64;
  __shared__ __align__(32) _Float16 Qg[64 * TS];
  __shared__ __align__(32) _Float16 Kg[64 * TS];
  __shared__ float mk[64];

  int t = threadIdx.x, lane = t & 31, w = t >> 5, grp = lane >> 4, ln = lane & 15;

  stage_gated_tile(Qp, Mg, bh, b, h, qbase, t, 0, Qg);
  __syncthreads();
  v16h a0 = load_a_frag(Qg + (w * 16) * TS, TS, lane, 0);
  v16h a1 = load_a_frag(Qg + (w * 16) * TS, TS, lane, 32);

  float mrun[8], lrun[8];
  #pragma unroll
  for (int r = 0; r < 8; ++r) { mrun[r] = -__builtin_inff(); lrun[r] = 0.f; }

  for (int kc = 0; kc < SS; kc += 64) {
    __syncthreads();
    stage_gated_tile(Kp, Mg, bh, b, h, kc, t, 1, Kg);
    if (t < 64) mk[t] = mask[b * SS + kc + t] * -1e9f;
    __syncthreads();

    float lv[4][8];
    #pragma unroll
    for (int j = 0; j < 4; ++j) {
      v8f acc = {0.f,0.f,0.f,0.f,0.f,0.f,0.f,0.f};
      v16h b0 = load_b_frag(Kg + (j * 16) * TS, TS, lane, 0);
      v16h b1 = load_b_frag(Kg + (j * 16) * TS, TS, lane, 32);
      acc = wmma_f16(a0, b0, acc);
      acc = wmma_f16(a1, b1, acc);
      float mval = mk[j * 16 + ln];
      #pragma unroll
      for (int r = 0; r < 8; ++r) lv[j][r] = acc[r] * SCALE + mval;
    }
    #pragma unroll
    for (int r = 0; r < 8; ++r) {
      float tm = fmaxf(fmaxf(lv[0][r], lv[1][r]), fmaxf(lv[2][r], lv[3][r]));
      tm = redmax16(tm);
      float mnew = fmaxf(mrun[r], tm);
      float ps = __expf(lv[0][r] - mnew) + __expf(lv[1][r] - mnew) +
                 __expf(lv[2][r] - mnew) + __expf(lv[3][r] - mnew);
      ps = redsum16(ps);
      lrun[r] = lrun[r] * __expf(mrun[r] - mnew) + ps;
      mrun[r] = mnew;
    }
  }
  if (ln == 0) {
    #pragma unroll
    for (int r = 0; r < 8; ++r) {
      int qr = qbase + w * 16 + grp * 8 + r;
      size_t idx = (size_t)bh * SS + qr;
      rowM[idx] = mrun[r];
      rowL[idx] = lrun[r];
    }
  }
}

// ============================================================================
// Kernel 3b: recompute logits, write normalized attn ONCE (512MB, write-only,
// non-temporal so it doesn't flush the K/V working set from L2), and fuse
// ctx = attn @ V with WMMA in the same pass (attn never re-read).
// ============================================================================
__global__ __launch_bounds__(128)
void attn_emit(const float* __restrict__ Qp, const float* __restrict__ Kp,
               const float* __restrict__ Vp, const float* __restrict__ Mg,
               const float* __restrict__ mask, const float* __restrict__ rowM,
               const float* __restrict__ rowL,
               float* __restrict__ attn, float* __restrict__ ctx) {
  int bh = blockIdx.x;
  int b = bh >> 4, h = bh & 15;
  int qbase = blockIdx.y * 64;
  __shared__ __align__(32) _Float16 Qg[64 * TS];
  __shared__ __align__(32) _Float16 Kg[64 * TS];
  __shared__ __align__(32) _Float16 Vt[64 * TS];      // transposed: [depth][key]
  __shared__ __align__(32) _Float16 Pl[4][16 * TS];   // per-wave P tile
  __shared__ float mk[64];

  int t = threadIdx.x, lane = t & 31, w = t >> 5, grp = lane >> 4, ln = lane & 15;

  stage_gated_tile(Qp, Mg, bh, b, h, qbase, t, 0, Qg);
  __syncthreads();
  v16h a0 = load_a_frag(Qg + (w * 16) * TS, TS, lane, 0);
  v16h a1 = load_a_frag(Qg + (w * 16) * TS, TS, lane, 32);

  float mrow[8], linv[8];
  #pragma unroll
  for (int r = 0; r < 8; ++r) {
    int qr = qbase + w * 16 + grp * 8 + r;
    size_t idx = (size_t)bh * SS + qr;
    mrow[r] = rowM[idx];
    linv[r] = 1.f / rowL[idx];
  }
  v8f ctxacc[4];
  #pragma unroll
  for (int j = 0; j < 4; ++j) ctxacc[j] = {0.f,0.f,0.f,0.f,0.f,0.f,0.f,0.f};

  for (int kc = 0; kc < SS; kc += 64) {
    __syncthreads();
    stage_gated_tile(Kp, Mg, bh, b, h, kc, t, 1, Kg);
    { // stage V tile transposed -> Vt[d][key]; pack (key,key+1) pairs as b32 stores
      int kp = (t >> 2) * 2, d0 = (t & 3) * 16;
      const float4* s0 = (const float4*)(Vp + ((size_t)(b * SS + kc + kp))     * DM + h * DD + d0);
      const float4* s1 = (const float4*)(Vp + ((size_t)(b * SS + kc + kp + 1)) * DM + h * DD + d0);
      #pragma unroll
      for (int i = 0; i < 4; ++i) {
        float4 fa = s0[i], fb = s1[i];
        *(h2*)(Vt + (d0 + 4*i + 0) * TS + kp) = h2{(_Float16)fa.x, (_Float16)fb.x};
        *(h2*)(Vt + (d0 + 4*i + 1) * TS + kp) = h2{(_Float16)fa.y, (_Float16)fb.y};
        *(h2*)(Vt + (d0 + 4*i + 2) * TS + kp) = h2{(_Float16)fa.z, (_Float16)fb.z};
        *(h2*)(Vt + (d0 + 4*i + 3) * TS + kp) = h2{(_Float16)fa.w, (_Float16)fb.w};
      }
    }
    if (t < 64) mk[t] = mask[b * SS + kc + t] * -1e9f;
    __syncthreads();

    float lv[4][8];
    #pragma unroll
    for (int j = 0; j < 4; ++j) {
      v8f acc = {0.f,0.f,0.f,0.f,0.f,0.f,0.f,0.f};
      v16h b0 = load_b_frag(Kg + (j * 16) * TS, TS, lane, 0);
      v16h b1 = load_b_frag(Kg + (j * 16) * TS, TS, lane, 32);
      acc = wmma_f16(a0, b0, acc);
      acc = wmma_f16(a1, b1, acc);
      float mval = mk[j * 16 + ln];
      #pragma unroll
      for (int r = 0; r < 8; ++r) lv[j][r] = acc[r] * SCALE + mval;
    }
    // normalize, stream attn out (non-temporal), stash P tile in LDS (wave-private)
    #pragma unroll
    for (int j = 0; j < 4; ++j) {
      #pragma unroll
      for (int r = 0; r < 8; ++r) {
        float p = __expf(lv[j][r] - mrow[r]) * linv[r];
        int qr = qbase + w * 16 + grp * 8 + r;
        __builtin_nontemporal_store(p, &attn[(((size_t)bh) * SS + qr) * SS + kc + j * 16 + ln]);
        Pl[w][(r + grp * 8) * TS + j * 16 + ln] = (_Float16)p;
      }
    }
    v16h pa0 = load_a_frag(Pl[w], TS, lane, 0);
    v16h pa1 = load_a_frag(Pl[w], TS, lane, 32);
    #pragma unroll
    for (int j = 0; j < 4; ++j) {
      v16h bv0 = load_b_frag(Vt + (j * 16) * TS, TS, lane, 0);
      v16h bv1 = load_b_frag(Vt + (j * 16) * TS, TS, lane, 32);
      ctxacc[j] = wmma_f16(pa0, bv0, ctxacc[j]);
      ctxacc[j] = wmma_f16(pa1, bv1, ctxacc[j]);
    }
  }
  // write ctx in (B,S,H*D) layout so the output projection is a plain GEMM
  #pragma unroll
  for (int j = 0; j < 4; ++j) {
    #pragma unroll
    for (int r = 0; r < 8; ++r) {
      int qr = qbase + w * 16 + grp * 8 + r;
      int dcol = j * 16 + ln;
      ctx[((size_t)(b * SS + qr)) * DM + h * DD + dcol] = ctxacc[j][r];
    }
  }
}

// ============================================================================
// host-side orchestration
// ============================================================================
extern "C" void kernel_launch(void* const* d_in, const int* in_sizes, int n_in,
                              void* d_out, int out_size, void* d_ws, size_t ws_size,
                              hipStream_t stream) {
  const float* d_v   = (const float*)d_in[0];
  const float* d_k   = (const float*)d_in[1];
  const float* d_q   = (const float*)d_in[2];
  const float* d_msk = (const float*)d_in[3];
  const float* Wq = (const float*)d_in[4];   const float* bq = (const float*)d_in[5];
  const float* Wk = (const float*)d_in[6];   const float* bk = (const float*)d_in[7];
  const float* Wv = (const float*)d_in[8];   const float* bv = (const float*)d_in[9];
  const float* Wgk = (const float*)d_in[10]; const float* bgk = (const float*)d_in[11];
  const float* Wgq = (const float*)d_in[12]; const float* bgq = (const float*)d_in[13];
  const float* Wm  = (const float*)d_in[14]; const float* bm  = (const float*)d_in[15];
  const float* Wo  = (const float*)d_in[16]; const float* bo  = (const float*)d_in[17];

  float* ws = (float*)d_ws;
  float* Qp   = ws;                 // 4096x1024
  float* Kp   = ws + 4194304;       // 4096x1024
  float* Vp   = ws + 8388608;       // 4096x1024
  float* Mg   = ws + 12582912;      // (B,H,S,2)
  float* rowM = ws + 12713984;      // (B,H,S)
  float* rowL = ws + 12779520;      // (B,H,S)
  float* PA   = ws + 12845056;      // 2x64x64
  float* Ug   = ws + 12853248;      // 2x64
  float* Vg   = ws + 12853376;      // 2x64
  float* Wg   = ws + 12853504;      // 2
  float* ctx  = ws + 12853760;      // 4096x1024

  float* outp = (float*)d_out;          // (B,S,1024)
  float* attn = outp + 4194304;         // (B,H,S,S)

  dim3 gemm_grid(DM / 64, (BB * SS) / 64);   // (16, 64)

  gemm_bias<<<gemm_grid, 128, 0, stream>>>(d_q, Wq, bq, Qp, BB * SS, DM, DM);
  gemm_bias<<<gemm_grid, 128, 0, stream>>>(d_k, Wk, bk, Kp, BB * SS, DM, DM);
  gemm_bias<<<gemm_grid, 128, 0, stream>>>(d_v, Wv, bv, Vp, BB * SS, DM, DM);

  gate_precompute<<<dim3(64, 2), 64, 0, stream>>>(Wgk, bgk, Wgq, bgq, Wm, bm, PA, Ug, Vg, Wg);
  gate_kernel<<<(BB * HH * SS) / 256, 256, 0, stream>>>(Qp, Kp, PA, Ug, Vg, Wg, Mg);

  attn_stats<<<dim3(BB * HH, SS / 64), 128, 0, stream>>>(Qp, Kp, Mg, d_msk, rowM, rowL);
  attn_emit<<<dim3(BB * HH, SS / 64), 128, 0, stream>>>(Qp, Kp, Vp, Mg, d_msk, rowM, rowL,
                                                        attn, ctx);

  gemm_bias<<<gemm_grid, 128, 0, stream>>>(ctx, Wo, bo, outp, BB * SS, DM, DM);
}
